// OurGMNCustom_8924942041965
// MI455X (gfx1250) — matrix operand order repaired and checked by reading, not compiled, not run on hardware
//
#include <hip/hip_runtime.h>
#include <hip/hip_bf16.h>
#include <math.h>

typedef __attribute__((ext_vector_type(16))) __bf16 v16bf;
typedef __attribute__((ext_vector_type(8)))  __bf16 v8bf;
typedef __attribute__((ext_vector_type(8)))  float  v8f;
typedef __attribute__((ext_vector_type(4)))  int    v4i;

#if defined(__HIP_DEVICE_COMPILE__) && __has_builtin(__builtin_amdgcn_global_load_async_to_lds_b128)
#define USE_ASYNC_LDS 1
// param0: v4i in global (addrspace 1); param1: LDS (addrspace 3)
#define ASYNC_CP16(gp, lp, off)                                              \
    __builtin_amdgcn_global_load_async_to_lds_b128(                          \
        (__attribute__((address_space(1))) v4i*)(gp),                        \
        (__attribute__((address_space(3))) v4i*)(lp), (off), 0)
#else
#define USE_ASYNC_LDS 0
#endif

__device__ inline void wait_async_lds() {
#if USE_ASYNC_LDS
#if __has_builtin(__builtin_amdgcn_s_wait_asynccnt)
    __builtin_amdgcn_s_wait_asynccnt(0);
#else
    asm volatile("s_wait_asynccnt 0x0" ::: "memory");
#endif
#endif
}

// ---------------- utility kernels ----------------

__global__ void k_fill(float* __restrict__ p, float v, int n) {
    int i = blockIdx.x * blockDim.x + threadIdx.x;
    if (i < n) p[i] = v;
}

__global__ void k_cvt_bf16(const float* __restrict__ s, __bf16* __restrict__ d, int n) {
    int i = blockIdx.x * blockDim.x + threadIdx.x;
    if (i < n) d[i] = (__bf16)s[i];
}

// W[K][N] fp32 (row-major) -> WT[N][K] bf16 (row-major): a lane's K-run is contiguous.
__global__ void k_cvt_wT(const float* __restrict__ W, __bf16* __restrict__ WT, int K, int N) {
    int i = blockIdx.x * blockDim.x + threadIdx.x;
    if (i >= K * N) return;
    int k = i / N, n = i - k * N;
    WT[(size_t)n * K + k] = (__bf16)W[i];
}

__device__ inline float atomic_max_f32(float* addr, float v) {
    if (v >= 0.f) return __int_as_float(atomicMax((int*)addr, __float_as_int(v)));
    return __uint_as_float(atomicMin((unsigned int*)addr, __float_as_uint(v)));
}

// ---------------- dense GEMM:  out = act(A[M,K]bf16 @ W[K,128] + b) ----------------
// block = 128 threads (4 waves). wave w: 16-row M-tile, all 8 N-tiles. grid.x = M/64.
__global__ __launch_bounds__(128)
void k_gemm_bias_act(const __bf16* __restrict__ A, int K,
                     const __bf16* __restrict__ WT,      // [128][K] bf16
                     const float* __restrict__ bias,     // [128]
                     float* __restrict__ outf,           // [M][128] or null
                     __bf16* __restrict__ outb,          // [M][128] or null
                     int act)                            // 1 = ELU
{
    const int lane  = threadIdx.x & 31;
    const int wave  = threadIdx.x >> 5;
    const int mBase = blockIdx.x * 64 + wave * 16;
    const int hiHalf = lane >> 4;          // 0: lanes 0-15, 1: lanes 16-31
    const int aSel = hiHalf ? 16 : 0;      // A frag k-sub-base (contiguous k = 16h + s)
    const int bSel = hiHalf ? 16 : 0;      // B frag k-sub-base per 32x16 layout
    const int ncol = lane & 15;
    const int mrow = mBase + ncol;

    v8f c[8];
#pragma unroll
    for (int t = 0; t < 8; ++t)
#pragma unroll
        for (int r = 0; r < 8; ++r) c[t][r] = 0.f;

    for (int k0 = 0; k0 < K; k0 += 32) {
        const __bf16* ap = A + (size_t)mrow * K + (k0 + aSel);
        v8bf alo = *(const v8bf*)(ap);
        v8bf ahi = *(const v8bf*)(ap + 8);
        v16bf a;
#pragma unroll
        for (int i = 0; i < 8; ++i) { a[i] = alo[i]; a[8 + i] = ahi[i]; }
#pragma unroll
        for (int t = 0; t < 8; ++t) {
            const __bf16* bp = WT + (size_t)(t * 16 + ncol) * K + (k0 + bSel);
            v8bf blo = *(const v8bf*)(bp);
            v8bf bhi = *(const v8bf*)(bp + 8);
            v16bf b;
#pragma unroll
            for (int i = 0; i < 8; ++i) { b[i] = blo[i]; b[8 + i] = bhi[i]; }
            c[t] = __builtin_amdgcn_wmma_f32_16x16x32_bf16(false, a, false, b,
                                                           (short)0, c[t], false, false);
        }
    }

    const int rbase = hiHalf ? 8 : 0;
#pragma unroll
    for (int t = 0; t < 8; ++t) {
        const int col = t * 16 + ncol;
        const float bb = bias ? bias[col] : 0.f;
#pragma unroll
        for (int r = 0; r < 8; ++r) {
            const int row = mBase + rbase + r;
            float v = c[t][r] + bb;
            if (act) v = (v > 0.f) ? v : expm1f(v);
            if (outf) outf[(size_t)row * 128 + col] = v;
            if (outb) outb[(size_t)row * 128 + col] = (__bf16)v;
        }
    }
}

// ---------------- column mean (Q summary) ----------------
__global__ void k_col_mean(const float* __restrict__ X, float* __restrict__ out, int N) {
    __shared__ float red[256];
    const int col = blockIdx.x;
    float s = 0.f;
    for (int r = threadIdx.x; r < N; r += 256) s += X[(size_t)r * 128 + col];
    red[threadIdx.x] = s;
    __syncthreads();
    for (int w = 128; w > 0; w >>= 1) {
        if (threadIdx.x < w) red[threadIdx.x] += red[threadIdx.x + w];
        __syncthreads();
    }
    if (threadIdx.x == 0) out[col] = red[0] / (float)N;
}

// ---------------- cross-edge logits ----------------
__global__ void k_cross_logits(const float* __restrict__ Xa, const float* __restrict__ Xb,
                               const int* __restrict__ src, const int* __restrict__ dst,
                               float* __restrict__ lg, int E) {
    int e = blockIdx.x * blockDim.x + threadIdx.x;
    if (e >= E) return;
    const float* pa = Xa + (size_t)src[e] * 128;
    const float* pb = Xb + (size_t)dst[e] * 128;
    float s = 0.f;
#pragma unroll 4
    for (int k = 0; k < 128; k += 4) {
        float4 a = *(const float4*)(pa + k);
        float4 b = *(const float4*)(pb + k);
        s += a.x * b.x + a.y * b.y + a.z * b.z + a.w * b.w;
    }
    lg[e] = s;
}

// ---------------- segment softmax pieces ----------------
__global__ void k_seg_max(const float* __restrict__ lg, const int* __restrict__ idx,
                          float* __restrict__ m, int E) {
    int e = blockIdx.x * blockDim.x + threadIdx.x;
    if (e >= E) return;
    atomic_max_f32(&m[idx[e]], lg[e]);
}

__global__ void k_seg_sum(const float* __restrict__ lg, const int* __restrict__ idx,
                          const float* __restrict__ m, float* __restrict__ s, int E) {
    int e = blockIdx.x * blockDim.x + threadIdx.x;
    if (e >= E) return;
    int i = idx[e];
    atomicAdd(&s[i], expf(lg[e] - m[i]));
}

// ---------------- cross-attention scatter: one wave per edge ----------------
__global__ __launch_bounds__(256)
void k_cross_scatter(const float* __restrict__ lg,
                     const int* __restrict__ src, const int* __restrict__ dst,
                     const float* __restrict__ mu, const float* __restrict__ su,
                     const float* __restrict__ mv, const float* __restrict__ sv,
                     const float* __restrict__ Xtv, const float* __restrict__ Xqv,
                     float* __restrict__ Xt2q, float* __restrict__ Xq2t, int E) {
    const int wid  = (blockIdx.x * 256 + threadIdx.x) >> 5;
    const int lane = threadIdx.x & 31;
    if (wid >= E) return;
    const int u = src[wid], v = dst[wid];
    const float l = lg[wid];
    const float wu = expf(l - mu[u]) / su[u];
    const float wv = expf(l - mv[v]) / sv[v];
    const int d0 = lane * 4;
    float4 tv = *(const float4*)(Xtv + (size_t)u * 128 + d0);  // values gathered by u (faithful)
    float4 qv = *(const float4*)(Xqv + (size_t)u * 128 + d0);
    float* pt = Xt2q + (size_t)u * 128 + d0;
    float* pq = Xq2t + (size_t)v * 128 + d0;
    atomicAdd(pt + 0, wu * tv.x); atomicAdd(pt + 1, wu * tv.y);
    atomicAdd(pt + 2, wu * tv.z); atomicAdd(pt + 3, wu * tv.w);
    atomicAdd(pq + 0, wv * qv.x); atomicAdd(pq + 1, wv * qv.y);
    atomicAdd(pq + 2, wv * qv.z); atomicAdd(pq + 3, wv * qv.w);
}

// ---------------- concat builders (fp32 -> bf16) ----------------
__global__ void k_cat2(const float* __restrict__ A, const float* __restrict__ B,
                       __bf16* __restrict__ D, int total) {  // [N][256]
    int i = blockIdx.x * blockDim.x + threadIdx.x;
    if (i >= total) return;
    int r = i >> 8, c = i & 255;
    float v = (c < 128) ? A[(size_t)r * 128 + c] : B[(size_t)r * 128 + (c - 128)];
    D[i] = (__bf16)v;
}

__global__ void k_cat3(const float* __restrict__ A, const float* __restrict__ B,
                       const float* __restrict__ Q, __bf16* __restrict__ D, int total) { // [N][384]
    int i = blockIdx.x * blockDim.x + threadIdx.x;
    if (i >= total) return;
    int r = i / 384, c = i - r * 384;
    float v = (c < 128) ? A[(size_t)r * 128 + c]
            : (c < 256) ? B[(size_t)r * 128 + (c - 128)]
                        : Q[c - 256];
    D[i] = (__bf16)v;
}

// ---------------- intra-edge attention logit:  [M[s],M[d]] . W_a + b ----------------
__global__ void k_edge_logit(const __bf16* __restrict__ M,
                             const int* __restrict__ src, const int* __restrict__ dst,
                             const float* __restrict__ Wa, const float* __restrict__ ba,
                             float* __restrict__ lg, int E) {
    int e = blockIdx.x * blockDim.x + threadIdx.x;
    if (e >= E) return;
    const __bf16* ps = M + (size_t)src[e] * 128;
    const __bf16* pd = M + (size_t)dst[e] * 128;
    float s = ba[0];
#pragma unroll 4
    for (int k = 0; k < 128; ++k)
        s += (float)ps[k] * Wa[k] + (float)pd[k] * Wa[128 + k];
    lg[e] = s;
}

// ---------------- intra edge message WMMA + softmax-weighted scatter ----------------
// 16 edges per block; A tile 16x256 bf16 in LDS (row stride 264 halfs, conflict-free).
// A staged with GLOBAL_LOAD_ASYNC_TO_LDS_B128 (ASYNCcnt) when available.
// 4 waves x 2 N-tiles x (256/32) k-steps of v_wmma_f32_16x16x32_bf16.
// E must be a multiple of 16 (holds: 262144).
__global__ __launch_bounds__(128)
void k_edge_wmma(const __bf16* __restrict__ M,
                 const int* __restrict__ src, const int* __restrict__ dst,
                 const float* __restrict__ lg,
                 const float* __restrict__ m, const float* __restrict__ s,
                 const __bf16* __restrict__ WT,      // [128][256] bf16 (W_v transposed)
                 const float* __restrict__ bias,     // [128]
                 float* __restrict__ out, int E) {
    __shared__ __align__(16) __bf16 Albs[16 * 264];
    __shared__ float aW[16];
    __shared__ int   dN[16];

    const int tid = threadIdx.x;
    const int e0  = blockIdx.x * 16;

    // hide dependent-gather latency of the NEXT block's scalars (global_prefetch_b8)
    if (tid == 0 && e0 + 16 < E) {
        __builtin_prefetch(src + e0 + 16, 0, 1);
        __builtin_prefetch(dst + e0 + 16, 0, 1);
        __builtin_prefetch(lg + e0 + 16, 0, 1);
    }

    {   // stage A: row r of the tile = [M[src[e0+r]] , M[dst[e0+r]]]
        const int r = tid >> 3, p = tid & 7;
        const int e = e0 + r;
        const int node = (p < 4) ? src[e] : dst[e];
        const int coff = (p & 3) * 32;
        const __bf16* gp = M + (size_t)node * 128 + coff;
        __bf16* lp = Albs + r * 264 + ((p < 4) ? 0 : 128) + coff;
#if USE_ASYNC_LDS
        // 64 B per thread, same stride both sides -> shared INST_OFFSET works.
        ASYNC_CP16(gp, lp, 0);
        ASYNC_CP16(gp, lp, 16);
        ASYNC_CP16(gp, lp, 32);
        ASYNC_CP16(gp, lp, 48);
#else
#pragma unroll
        for (int i = 0; i < 4; ++i) *(v8bf*)(lp + i * 8) = *(const v8bf*)(gp + i * 8);
#endif
    }
    if (tid < 16) {  // softmax weight + destination, overlapped with async DMA
        const int e = e0 + tid;
        const int d = dst[e];
        dN[tid] = d;
        aW[tid] = expf(lg[e] - m[d]) / s[d];
    }
    wait_async_lds();
    __syncthreads();

    const int lane = tid & 31;
    const int wave = tid >> 5;
    const int hiHalf = lane >> 4;
    const int aSel = hiHalf ? 16 : 0;
    const int bSel = hiHalf ? 16 : 0;
    const int ncol = lane & 15;
    const int mrow = lane & 15;

    v8f c0, c1;
#pragma unroll
    for (int r = 0; r < 8; ++r) { c0[r] = 0.f; c1[r] = 0.f; }

#pragma unroll
    for (int k0 = 0; k0 < 256; k0 += 32) {
        const __bf16* ap = Albs + mrow * 264 + k0 + aSel;
        v8bf alo = *(const v8bf*)(ap);
        v8bf ahi = *(const v8bf*)(ap + 8);
        v16bf a;
#pragma unroll
        for (int i = 0; i < 8; ++i) { a[i] = alo[i]; a[8 + i] = ahi[i]; }

        const __bf16* bp0 = WT + (size_t)(wave * 32 + ncol) * 256 + (k0 + bSel);
        v8bf b0lo = *(const v8bf*)(bp0);
        v8bf b0hi = *(const v8bf*)(bp0 + 8);
        v16bf b0;
#pragma unroll
        for (int i = 0; i < 8; ++i) { b0[i] = b0lo[i]; b0[8 + i] = b0hi[i]; }
        c0 = __builtin_amdgcn_wmma_f32_16x16x32_bf16(false, a, false, b0, (short)0, c0, false, false);

        const __bf16* bp1 = WT + (size_t)(wave * 32 + 16 + ncol) * 256 + (k0 + bSel);
        v8bf b1lo = *(const v8bf*)(bp1);
        v8bf b1hi = *(const v8bf*)(bp1 + 8);
        v16bf b1;
#pragma unroll
        for (int i = 0; i < 8; ++i) { b1[i] = b1lo[i]; b1[8 + i] = b1hi[i]; }
        c1 = __builtin_amdgcn_wmma_f32_16x16x32_bf16(false, a, false, b1, (short)0, c1, false, false);
    }

    const int rbase = hiHalf ? 8 : 0;
    const int col0 = wave * 32 + ncol;
    const int col1 = col0 + 16;
    const float bb0 = bias[col0], bb1 = bias[col1];
#pragma unroll
    for (int r = 0; r < 8; ++r) {
        const int row = rbase + r;
        const float ar = aW[row];
        float* op = out + (size_t)dN[row] * 128;
        atomicAdd(op + col0, (c0[r] + bb0) * ar);
        atomicAdd(op + col1, (c1[r] + bb1) * ar);
    }
}

// ---------------- host orchestration ----------------
extern "C" void kernel_launch(void* const* d_in, const int* in_sizes, int n_in,
                              void* d_out, int out_size, void* d_ws, size_t ws_size,
                              hipStream_t stream) {
    (void)n_in; (void)ws_size;

    const float* Xq   = (const float*)d_in[0];
    const int*   eiq  = (const int*)d_in[1];
    const float* Xt   = (const float*)d_in[2];
    const int*   eit  = (const int*)d_in[3];
    const int*   csrc = (const int*)d_in[6];
    const int*   cdst = (const int*)d_in[7];
    // norm_q/norm_t/node_mask/only_run_inter unused by the reference math path.
    const float* W_ac_q = (const float*)d_in[10]; const float* b_ac_q = (const float*)d_in[11];
    const float* W_ac_t = (const float*)d_in[12]; const float* b_ac_t = (const float*)d_in[13];
    const float* W_vc_q = (const float*)d_in[14]; const float* b_vc_q = (const float*)d_in[15];
    const float* W_vc_t = (const float*)d_in[16]; const float* b_vc_t = (const float*)d_in[17];
    const float* W_mq   = (const float*)d_in[18]; const float* b_mq   = (const float*)d_in[19];
    const float* W_mt   = (const float*)d_in[20]; const float* b_mt   = (const float*)d_in[21];
    const float* W_aq   = (const float*)d_in[22]; const float* b_aq   = (const float*)d_in[23];
    const float* W_vq   = (const float*)d_in[24]; const float* b_vq   = (const float*)d_in[25];
    const float* W_at   = (const float*)d_in[26]; const float* b_at   = (const float*)d_in[27];
    const float* W_vt   = (const float*)d_in[28]; const float* b_vt   = (const float*)d_in[29];

    const int NQ = in_sizes[0] / 128;
    const int NT = in_sizes[2] / 128;
    const int EC = in_sizes[6];
    const int EQ = in_sizes[1] / 2;
    const int ET = in_sizes[3] / 2;

    // ---- workspace carve-up (256B aligned chunks) ----
    char* ws = (char*)d_ws;
    size_t off = 0;
    auto alloc = [&](size_t bytes) -> void* {
        void* p = ws + off;
        off += (bytes + 255) & ~(size_t)255;
        return p;
    };
    __bf16* Xq_bf  = (__bf16*)alloc((size_t)NQ * 128 * 2);
    __bf16* Xt_bf  = (__bf16*)alloc((size_t)NT * 128 * 2);
    __bf16* WT_acq = (__bf16*)alloc(128 * 128 * 2);
    __bf16* WT_act = (__bf16*)alloc(128 * 128 * 2);
    __bf16* WT_vcq = (__bf16*)alloc(128 * 128 * 2);
    __bf16* WT_vct = (__bf16*)alloc(128 * 128 * 2);
    __bf16* WT_mq  = (__bf16*)alloc(128 * 256 * 2);
    __bf16* WT_mt  = (__bf16*)alloc(128 * 384 * 2);
    __bf16* WT_vq  = (__bf16*)alloc(128 * 256 * 2);
    __bf16* WT_vt  = (__bf16*)alloc(128 * 256 * 2);
    float* Xq_ac = (float*)alloc((size_t)NQ * 128 * 4);
    float* Xt_ac = (float*)alloc((size_t)NT * 128 * 4);
    float* Xq_vc = (float*)alloc((size_t)NQ * 128 * 4);
    float* Xt_vc = (float*)alloc((size_t)NT * 128 * 4);
    float* lgc   = (float*)alloc((size_t)EC * 4);
    float* Qmean = (float*)alloc(128 * 4);
    __bf16* catq = (__bf16*)alloc((size_t)NQ * 256 * 2);
    __bf16* catt = (__bf16*)alloc((size_t)NT * 384 * 2);
    __bf16* Mq   = (__bf16*)alloc((size_t)NQ * 128 * 2);
    __bf16* Mt   = (__bf16*)alloc((size_t)NT * 128 * 2);
    float* lgq   = (float*)alloc((size_t)EQ * 4);
    float* lgt   = (float*)alloc((size_t)ET * 4);
    // contiguous zero-init region
    const int zN = NQ * 128 + NT * 128 + 2 * NQ + 2 * NT;
    float* zbase = (float*)alloc((size_t)zN * 4);
    float* Xt2q = zbase;
    float* Xq2t = Xt2q + (size_t)NQ * 128;
    float* s_u  = Xq2t + (size_t)NT * 128;
    float* s_v  = s_u + NQ;
    float* s_iq = s_v + NT;
    float* s_it = s_iq + NQ;
    // contiguous -inf region
    const int nN = 2 * (NQ + NT);
    float* nbase = (float*)alloc((size_t)nN * 4);
    float* m_u  = nbase;
    float* m_v  = m_u + NQ;
    float* m_iq = m_v + NT;
    float* m_it = m_iq + NQ;

    float* outQ = (float*)d_out;
    float* outT = outQ + (size_t)NQ * 128;

    const int TB = 256;
    auto nb = [&](int n) { return (n + TB - 1) / TB; };

    // ---- init accumulators (d_out / d_ws are poisoned by the harness) ----
    k_fill<<<nb(zN), TB, 0, stream>>>(zbase, 0.f, zN);
    k_fill<<<nb(nN), TB, 0, stream>>>(nbase, -INFINITY, nN);
    k_fill<<<nb(out_size), TB, 0, stream>>>((float*)d_out, 0.f, out_size);

    // ---- bf16 conversions ----
    k_cvt_bf16<<<nb(NQ * 128), TB, 0, stream>>>(Xq, Xq_bf, NQ * 128);
    k_cvt_bf16<<<nb(NT * 128), TB, 0, stream>>>(Xt, Xt_bf, NT * 128);
    k_cvt_wT<<<nb(128 * 128), TB, 0, stream>>>(W_ac_q, WT_acq, 128, 128);
    k_cvt_wT<<<nb(128 * 128), TB, 0, stream>>>(W_ac_t, WT_act, 128, 128);
    k_cvt_wT<<<nb(128 * 128), TB, 0, stream>>>(W_vc_q, WT_vcq, 128, 128);
    k_cvt_wT<<<nb(128 * 128), TB, 0, stream>>>(W_vc_t, WT_vct, 128, 128);
    k_cvt_wT<<<nb(256 * 128), TB, 0, stream>>>(W_mq, WT_mq, 256, 128);
    k_cvt_wT<<<nb(384 * 128), TB, 0, stream>>>(W_mt, WT_mt, 384, 128);
    k_cvt_wT<<<nb(256 * 128), TB, 0, stream>>>(W_vq, WT_vq, 256, 128);
    k_cvt_wT<<<nb(256 * 128), TB, 0, stream>>>(W_vt, WT_vt, 256, 128);

    // ---- projection GEMMs (+bias+ELU) via WMMA ----
    k_gemm_bias_act<<<NQ / 64, 128, 0, stream>>>(Xq_bf, 128, WT_acq, b_ac_q, Xq_ac, nullptr, 1);
    k_gemm_bias_act<<<NT / 64, 128, 0, stream>>>(Xt_bf, 128, WT_act, b_ac_t, Xt_ac, nullptr, 1);
    k_gemm_bias_act<<<NQ / 64, 128, 0, stream>>>(Xq_bf, 128, WT_vcq, b_vc_q, Xq_vc, nullptr, 1);
    k_gemm_bias_act<<<NT / 64, 128, 0, stream>>>(Xt_bf, 128, WT_vct, b_vc_t, Xt_vc, nullptr, 1);

    // ---- global query summary ----
    k_col_mean<<<128, 256, 0, stream>>>(Xq, Qmean, NQ);

    // ---- cross attention ----
    k_cross_logits<<<nb(EC), TB, 0, stream>>>(Xq_ac, Xt_ac, csrc, cdst, lgc, EC);
    k_seg_max<<<nb(EC), TB, 0, stream>>>(lgc, csrc, m_u, EC);
    k_seg_max<<<nb(EC), TB, 0, stream>>>(lgc, cdst, m_v, EC);
    k_seg_sum<<<nb(EC), TB, 0, stream>>>(lgc, csrc, m_u, s_u, EC);
    k_seg_sum<<<nb(EC), TB, 0, stream>>>(lgc, cdst, m_v, s_v, EC);
    k_cross_scatter<<<(EC * 32 + TB - 1) / TB, TB, 0, stream>>>(
        lgc, csrc, cdst, m_u, s_u, m_v, s_v, Xt_vc, Xq_vc, Xt2q, Xq2t, EC);

    // ---- merge GEMMs ----
    k_cat2<<<nb(NQ * 256), TB, 0, stream>>>(Xq, Xt2q, catq, NQ * 256);
    k_cat3<<<nb(NT * 384), TB, 0, stream>>>(Xt, Xq2t, Qmean, catt, NT * 384);
    k_gemm_bias_act<<<NQ / 64, 128, 0, stream>>>(catq, 256, WT_mq, b_mq, nullptr, Mq, 0);
    k_gemm_bias_act<<<NT / 64, 128, 0, stream>>>(catt, 384, WT_mt, b_mt, nullptr, Mt, 0);

    // ---- intra message passing: t-graph ----
    const int* src_t = eit;
    const int* dst_t = eit + ET;
    k_edge_logit<<<nb(ET), TB, 0, stream>>>(Mt, src_t, dst_t, W_at, b_at, lgt, ET);
    k_seg_max<<<nb(ET), TB, 0, stream>>>(lgt, dst_t, m_it, ET);
    k_seg_sum<<<nb(ET), TB, 0, stream>>>(lgt, dst_t, m_it, s_it, ET);
    k_edge_wmma<<<ET / 16, 128, 0, stream>>>(Mt, src_t, dst_t, lgt, m_it, s_it,
                                             WT_vt, b_vt, outT, ET);

    // ---- intra message passing: q-graph ----
    const int* src_q = eiq;
    const int* dst_q = eiq + EQ;
    k_edge_logit<<<nb(EQ), TB, 0, stream>>>(Mq, src_q, dst_q, W_aq, b_aq, lgq, EQ);
    k_seg_max<<<nb(EQ), TB, 0, stream>>>(lgq, dst_q, m_iq, EQ);
    k_seg_sum<<<nb(EQ), TB, 0, stream>>>(lgq, dst_q, m_iq, s_iq, EQ);
    k_edge_wmma<<<EQ / 16, 128, 0, stream>>>(Mq, src_q, dst_q, lgq, m_iq, s_iq,
                                             WT_vq, b_vq, outQ, EQ);
}
